// TopologyPreservingLoss_20710332302052
// MI455X (gfx1250) — compile-verified, matrix operand order checked by reading, not run on hardware
//
#include <hip/hip_runtime.h>
#include <hip/hip_bf16.h>

// ---------------------------------------------------------------------------
// Problem constants (from reference): B=2, C=4, D=64, H=128, W=128
// ---------------------------------------------------------------------------
#define DD 64
#define HH 128
#define WW 128
#define NVOX (DD * HH * WW)      // 1<<20
#define NMASK 12                 // 6 pred + 6 target masks
#define BIGL NVOX                // "infinity" label

// LDS tile for stencil relaxation
#define TZ 4
#define TY 16
#define TX 16
#define HZ (TZ + 2)
#define HY (TY + 2)
#define HX (TX + 2)
#define HALO_N (HZ * HY * HX)    // 1944
#define LOCAL_IT 8

#define ITERS_CC 48              // global sweeps for CC label propagation
#define ITERS_FL 32              // global sweeps for flood fill

typedef __attribute__((ext_vector_type(8))) int v8i;
typedef __attribute__((ext_vector_type(4))) int v4i;
typedef __attribute__((ext_vector_type(4))) unsigned int v4u;

#if __has_builtin(__builtin_amdgcn_tensor_load_to_lds) && \
    __has_builtin(__builtin_amdgcn_s_wait_tensorcnt)
#define USE_TDM 1
#else
#define USE_TDM 0
#endif

__device__ __forceinline__ int imin(int a, int b) { return a < b ? a : b; }
__device__ __forceinline__ int imax(int a, int b) { return a > b ? a : b; }
__device__ __forceinline__ int iabs(int a) { return a < 0 ? -a : a; }

#if USE_TDM
// ---------------------------------------------------------------------------
// TDM: stage one contiguous row of `nbytes` bytes from global into LDS.
// D# per CDNA5 ISA ch.8: group0 = {count, lds_addr, global_addr, type=2},
// group1 = 1-D descriptor: data_size=1B, tensor_dim0=tile_dim0=nbytes,
// tensor_dim1=1, tensor_dim0_stride=nbytes. Groups 2/3 unused (<=2D tensor).
// ---------------------------------------------------------------------------
__device__ __forceinline__ void tdm_row_to_lds(unsigned lds_addr,
                                               const unsigned char* g,
                                               unsigned nbytes) {
    unsigned long long ga = (unsigned long long)(size_t)g;
    v4u g0;
    g0[0] = 1u;                                              // count=1
    g0[1] = lds_addr;                                        // lds_addr
    g0[2] = (unsigned)ga;                                    // global_addr lo
    g0[3] = (unsigned)((ga >> 32) & 0x01FFFFFFu) | (2u << 30); // hi + type=2
    v8i g1 = {};
    g1[1] = (int)(nbytes << 16);                 // tensor_dim0[15:0] @ bits63:48
    g1[2] = (int)((nbytes >> 16) | (1u << 16));  // tensor_dim0 hi, tensor_dim1=1
    g1[3] = (int)(nbytes << 16);                 // tile_dim0 @ bits127:112
    g1[5] = (int)nbytes;                         // tensor_dim0_stride lo32
    v4i z4 = {};
#if __clang_major__ >= 23
    v8i z8 = {};
    __builtin_amdgcn_tensor_load_to_lds(g0, g1, z4, z4, z8, 0);
#else
    __builtin_amdgcn_tensor_load_to_lds(g0, g1, z4, z4, 0);
#endif
}
#endif

// ---------------------------------------------------------------------------
// 0. zero the 36 stats slots (roots[12], rootsE[12], cav[12])
// ---------------------------------------------------------------------------
__global__ void zero_stats(int* __restrict__ st) {
    if (threadIdx.x < 36) st[threadIdx.x] = 0;
}

// ---------------------------------------------------------------------------
// 1. softmax-threshold pred masks + target equality masks
// ---------------------------------------------------------------------------
__global__ __launch_bounds__(256)
void mask_kernel(const float* __restrict__ pred, const int* __restrict__ tgt,
                 unsigned char* __restrict__ mask) {
    int g = blockIdx.x * 256 + threadIdx.x;     // over B*NVOX
    if (g >= 2 * NVOX) return;
    int b = g >> 20;
    int v = g & (NVOX - 1);
    const float* p = pred + ((size_t)b * 4) * (size_t)NVOX + v;
    float x0 = p[0];
    float x1 = p[(size_t)NVOX];
    float x2 = p[(size_t)2 * NVOX];
    float x3 = p[(size_t)3 * NVOX];
    float mx = fmaxf(fmaxf(x0, x1), fmaxf(x2, x3));
    float e0 = __expf(x0 - mx), e1 = __expf(x1 - mx);
    float e2 = __expf(x2 - mx), e3 = __expf(x3 - mx);
    float h = 0.5f * (e0 + e1 + e2 + e3);
    int t = tgt[(size_t)b * NVOX + v];
    size_t pb = ((size_t)b * 3) * (size_t)NVOX + v;
    mask[pb]                    = (unsigned char)(e1 > h);
    mask[pb + (size_t)NVOX]     = (unsigned char)(e2 > h);
    mask[pb + (size_t)2 * NVOX] = (unsigned char)(e3 > h);
    size_t tb = ((size_t)(6 + b * 3)) * (size_t)NVOX + v;
    mask[tb]                    = (unsigned char)(t == 1);
    mask[tb + (size_t)NVOX]     = (unsigned char)(t == 2);
    mask[tb + (size_t)2 * NVOX] = (unsigned char)(t == 3);
}

// ---------------------------------------------------------------------------
// 2. binary erosion (6-connectivity, border false)
// ---------------------------------------------------------------------------
__global__ __launch_bounds__(256)
void erode_kernel(const unsigned char* __restrict__ mask,
                  unsigned char* __restrict__ ero) {
    int g = blockIdx.x * 256 + threadIdx.x;     // over 12*NVOX
    int v = g & (NVOX - 1);
    int z = v >> 14;            // HH*WW = 1<<14
    int y = (v >> 7) & 127;
    int x = v & 127;
    const unsigned char* M = mask + ((size_t)(g >> 20) << 20);
    unsigned char c = M[v];
    unsigned char a = c;
    a &= (z > 0)      ? M[v - HH * WW] : (unsigned char)0;
    a &= (z < DD - 1) ? M[v + HH * WW] : (unsigned char)0;
    a &= (y > 0)      ? M[v - WW]      : (unsigned char)0;
    a &= (y < HH - 1) ? M[v + WW]      : (unsigned char)0;
    a &= (x > 0)      ? M[v - 1]       : (unsigned char)0;
    a &= (x < WW - 1) ? M[v + 1]       : (unsigned char)0;
    ero[g] = a;
}

// ---------------------------------------------------------------------------
// 3. label init: lbl = mask ? voxel_index : BIG
// ---------------------------------------------------------------------------
__global__ __launch_bounds__(256)
void lbl_init(const unsigned char* __restrict__ mask, int* __restrict__ lbl) {
    int g = blockIdx.x * 256 + threadIdx.x;     // over 12*NVOX
    lbl[g] = mask[g] ? (g & (NVOX - 1)) : BIGL;
}

// ---------------------------------------------------------------------------
// 4. LDS-tiled min-label relaxation: 8 local sub-iterations per launch.
//    Monotone (min) operator -> tile-local relaxation converges to the same
//    fixpoint as the reference Jacobi while-loop.
// ---------------------------------------------------------------------------
__global__ __launch_bounds__(256)
void cc_step(const int* __restrict__ src, int* __restrict__ dst,
             const unsigned char* __restrict__ mask) {
    __shared__ int ls[HZ][HY][HX];
    const int m  = blockIdx.z / (DD / TZ);
    const int z0 = (blockIdx.z % (DD / TZ)) * TZ;
    const int y0 = blockIdx.y * TY;
    const int x0 = blockIdx.x * TX;
    const size_t mb = (size_t)m << 20;
    const int tid = threadIdx.x;

    // halo box load (OOB -> BIG)
    for (int h = tid; h < HALO_N; h += 256) {
        int lz = h / (HY * HX);
        int r  = h % (HY * HX);
        int ly = r / HX;
        int lx = r % HX;
        int gz = z0 + lz - 1, gy = y0 + ly - 1, gx = x0 + lx - 1;
        int val = BIGL;
        if (gz >= 0 && gz < DD && gy >= 0 && gy < HH && gx >= 0 && gx < WW)
            val = src[mb + (size_t)((gz * HH + gy) * WW + gx)];
        ls[lz][ly][lx] = val;
    }
    int tz[4], ty[4], tx[4];
    unsigned char mm[4];
    #pragma unroll
    for (int k = 0; k < 4; ++k) {
        int i = tid + k * 256;
        tz[k] = i / (TY * TX);
        ty[k] = (i / TX) % TY;
        tx[k] = i % TX;
        mm[k] = mask[mb + (size_t)(((z0 + tz[k]) * HH + (y0 + ty[k])) * WW + (x0 + tx[k]))];
    }
    __syncthreads();
    #pragma unroll 1
    for (int it = 0; it < LOCAL_IT; ++it) {
        int nv[4];
        #pragma unroll
        for (int k = 0; k < 4; ++k) {
            int lz = tz[k] + 1, ly = ty[k] + 1, lx = tx[k] + 1;
            int a = imin(ls[lz - 1][ly][lx], ls[lz + 1][ly][lx]);
            int b = imin(ls[lz][ly - 1][lx], ls[lz][ly + 1][lx]);
            int c = imin(ls[lz][ly][lx - 1], ls[lz][ly][lx + 1]);
            int mn = imin(imin(a, b), imin(c, ls[lz][ly][lx]));
            nv[k] = mm[k] ? mn : BIGL;
        }
        __syncthreads();
        #pragma unroll
        for (int k = 0; k < 4; ++k)
            ls[tz[k] + 1][ty[k] + 1][tx[k] + 1] = nv[k];
        __syncthreads();
    }
    #pragma unroll
    for (int k = 0; k < 4; ++k)
        dst[mb + (size_t)(((z0 + tz[k]) * HH + (y0 + ty[k])) * WW + (x0 + tx[k]))] =
            ls[tz[k] + 1][ty[k] + 1][tx[k] + 1];
}

// ---------------------------------------------------------------------------
// 5. flood fill init: reach = background & border
// ---------------------------------------------------------------------------
__global__ __launch_bounds__(256)
void flood_init(const unsigned char* __restrict__ mask,
                unsigned char* __restrict__ r) {
    int g = blockIdx.x * 256 + threadIdx.x;     // over 12*NVOX
    int v = g & (NVOX - 1);
    int z = v >> 14, y = (v >> 7) & 127, x = v & 127;
    int border = (z == 0) | (z == DD - 1) | (y == 0) | (y == HH - 1) |
                 (x == 0) | (x == WW - 1);
    r[g] = (unsigned char)((mask[g] == 0) & border);
}

// ---------------------------------------------------------------------------
// 6. LDS-tiled OR flood relaxation (background reachability)
// ---------------------------------------------------------------------------
__global__ __launch_bounds__(256)
void flood_step(const unsigned char* __restrict__ src,
                unsigned char* __restrict__ dst,
                const unsigned char* __restrict__ mask) {
    __shared__ unsigned char ls[HZ][HY][HX];
    const int m  = blockIdx.z / (DD / TZ);
    const int z0 = (blockIdx.z % (DD / TZ)) * TZ;
    const int y0 = blockIdx.y * TY;
    const int x0 = blockIdx.x * TX;
    const size_t mb = (size_t)m << 20;
    const int tid = threadIdx.x;

    for (int h = tid; h < HALO_N; h += 256) {
        int lz = h / (HY * HX);
        int r  = h % (HY * HX);
        int ly = r / HX;
        int lx = r % HX;
        int gz = z0 + lz - 1, gy = y0 + ly - 1, gx = x0 + lx - 1;
        unsigned char val = 0;
        if (gz >= 0 && gz < DD && gy >= 0 && gy < HH && gx >= 0 && gx < WW)
            val = src[mb + (size_t)((gz * HH + gy) * WW + gx)];
        ls[lz][ly][lx] = val;
    }
    int tz[4], ty[4], tx[4];
    unsigned char bg[4];
    #pragma unroll
    for (int k = 0; k < 4; ++k) {
        int i = tid + k * 256;
        tz[k] = i / (TY * TX);
        ty[k] = (i / TX) % TY;
        tx[k] = i % TX;
        bg[k] = (unsigned char)(mask[mb + (size_t)(((z0 + tz[k]) * HH + (y0 + ty[k])) * WW + (x0 + tx[k]))] == 0);
    }
    __syncthreads();
    #pragma unroll 1
    for (int it = 0; it < LOCAL_IT; ++it) {
        unsigned char nv[4];
        #pragma unroll
        for (int k = 0; k < 4; ++k) {
            int lz = tz[k] + 1, ly = ty[k] + 1, lx = tx[k] + 1;
            unsigned char o = (unsigned char)(ls[lz][ly][lx] |
                ls[lz - 1][ly][lx] | ls[lz + 1][ly][lx] |
                ls[lz][ly - 1][lx] | ls[lz][ly + 1][lx] |
                ls[lz][ly][lx - 1] | ls[lz][ly][lx + 1]);
            nv[k] = (unsigned char)(bg[k] & o);
        }
        __syncthreads();
        #pragma unroll
        for (int k = 0; k < 4; ++k)
            ls[tz[k] + 1][ty[k] + 1][tx[k] + 1] = nv[k];
        __syncthreads();
    }
    #pragma unroll
    for (int k = 0; k < 4; ++k)
        dst[mb + (size_t)(((z0 + tz[k]) * HH + (y0 + ty[k])) * WW + (x0 + tx[k]))] =
            ls[tz[k] + 1][ty[k] + 1][tx[k] + 1];
}

// ---------------------------------------------------------------------------
// 7. WMMA-IU8 reductions. Each wave owns 1024-voxel chunks: the mask bytes
//    for a chunk are staged into a private LDS slot by the Tensor Data Mover
//    (s_wait_tensorcnt for completion), labels/reach stream in as b128 loads,
//    predicates are packed word-wise into an 8-VGPR A operand, and
//    V_WMMA_I32_16X16X64_IU8 against an all-ones B accumulates the count
//    (all 16 output columns identical -> total = matrix-sum >> 4).
//    Loop trip counts are wave-uniform -> EXEC all ones at the WMMA.
// ---------------------------------------------------------------------------
__global__ __launch_bounds__(256)
void reduce_roots(const int* __restrict__ lbl,
                  const unsigned char* __restrict__ mask,
                  int* __restrict__ out) {
    const int m    = blockIdx.y;
    const int lane = threadIdx.x & 31;
    const int wave = threadIdx.x >> 5;
    const int gwave  = blockIdx.x * 8 + wave;
    const int nwaves = gridDim.x * 8;
    const int* L = lbl + ((size_t)m << 20);
    const unsigned char* M = mask + ((size_t)m << 20);

#if USE_TDM
    __shared__ __align__(16) unsigned char stage[8 * 1024];
    unsigned slot = (unsigned)(size_t)&stage[wave * 1024];
#endif

    v8i ones;
    #pragma unroll
    for (int i = 0; i < 8; ++i) ones[i] = 0x01010101;
    v8i acc = {};

    const int nchunk = NVOX / 1024;
    for (int c = gwave; c < nchunk; c += nwaves) {
        int base = c * 1024 + lane * 32;
        __builtin_prefetch(L + base + 1024, 0, 0);   // global_prefetch_b8
#if USE_TDM
        tdm_row_to_lds(slot, M + c * 1024, 1024);    // stage mask chunk via TDM
        __builtin_amdgcn_s_wait_tensorcnt(0);
        const unsigned* Mw = (const unsigned*)&stage[wave * 1024 + lane * 32];
#else
        const unsigned* Mw = (const unsigned*)(M + base);
#endif
        const int4* Lp = (const int4*)(L + base);    // 128-bit label loads
        v8i a;
        #pragma unroll
        for (int r = 0; r < 8; ++r) {
            int4 l = Lp[r];
            int v = base + r * 4;
            unsigned e = (unsigned)(l.x == v) |
                         ((unsigned)(l.y == v + 1) << 8) |
                         ((unsigned)(l.z == v + 2) << 16) |
                         ((unsigned)(l.w == v + 3) << 24);
            a[r] = (int)(e & Mw[r]);   // mask bytes are 0/1
        }
        acc = __builtin_amdgcn_wmma_i32_16x16x64_iu8(false, a, false, ones, acc,
                                                     false, false);
    }
    int s = 0;
    #pragma unroll
    for (int i = 0; i < 8; ++i) s += acc[i];
    #pragma unroll
    for (int off = 16; off > 0; off >>= 1) s += __shfl_xor(s, off, 32);
    if (lane == 0) atomicAdd(out + m, s >> 4);
}

__global__ __launch_bounds__(256)
void reduce_cavity(const unsigned char* __restrict__ reach,
                   const unsigned char* __restrict__ mask,
                   int* __restrict__ out) {
    const int m    = blockIdx.y;
    const int lane = threadIdx.x & 31;
    const int wave = threadIdx.x >> 5;
    const int gwave  = blockIdx.x * 8 + wave;
    const int nwaves = gridDim.x * 8;
    const unsigned char* R = reach + ((size_t)m << 20);
    const unsigned char* M = mask + ((size_t)m << 20);

#if USE_TDM
    __shared__ __align__(16) unsigned char stage[8 * 1024];
    unsigned slot = (unsigned)(size_t)&stage[wave * 1024];
#endif

    v8i ones;
    #pragma unroll
    for (int i = 0; i < 8; ++i) ones[i] = 0x01010101;
    v8i acc = {};

    const int nchunk = NVOX / 1024;
    for (int c = gwave; c < nchunk; c += nwaves) {
        int base = c * 1024 + lane * 32;
        __builtin_prefetch(R + base + 1024, 0, 0);
#if USE_TDM
        tdm_row_to_lds(slot, M + c * 1024, 1024);
        __builtin_amdgcn_s_wait_tensorcnt(0);
        const unsigned* Mw = (const unsigned*)&stage[wave * 1024 + lane * 32];
#else
        const unsigned* Mw = (const unsigned*)(M + base);
#endif
        const unsigned* Rw = (const unsigned*)(R + base);
        v8i a;
        #pragma unroll
        for (int r = 0; r < 8; ++r) {
            // bytes are 0/1: cavity <=> mask==0 && reach==0
            a[r] = (int)((~(Mw[r] | Rw[r])) & 0x01010101u);
        }
        acc = __builtin_amdgcn_wmma_i32_16x16x64_iu8(false, a, false, ones, acc,
                                                     false, false);
    }
    int s = 0;
    #pragma unroll
    for (int i = 0; i < 8; ++i) s += acc[i];
    #pragma unroll
    for (int off = 16; off > 0; off >>= 1) s += __shfl_xor(s, off, 32);
    if (lane == 0) atomicAdd(out + m, s >> 4);
}

// ---------------------------------------------------------------------------
// 8. finalize: b0=roots, b1=max(0,b0-b0e), b2=cav//100, L1 diff, scale
// ---------------------------------------------------------------------------
__global__ void finalize(const int* __restrict__ st, float* __restrict__ out) {
    if (threadIdx.x == 0 && blockIdx.x == 0) {
        const int* roots  = st;
        const int* rootsE = st + 12;
        const int* cav    = st + 24;
        int acc = 0;
        for (int m = 0; m < 6; ++m) {
            int pb0 = roots[m];
            int pb1 = imax(0, roots[m] - rootsE[m]);
            int pb2 = cav[m] / 100;
            int tb0 = roots[6 + m];
            int tb1 = imax(0, roots[6 + m] - rootsE[6 + m]);
            int tb2 = cav[6 + m] / 100;
            acc += iabs(pb0 - tb0) + iabs(pb1 - tb1) + iabs(pb2 - tb2);
        }
        out[0] = 0.1f * (float)acc / 6.0f;
    }
}

// ---------------------------------------------------------------------------
// Launch orchestration (fixed sequence, graph-capture safe)
// ---------------------------------------------------------------------------
extern "C" void kernel_launch(void* const* d_in, const int* in_sizes, int n_in,
                              void* d_out, int out_size, void* d_ws, size_t ws_size,
                              hipStream_t stream) {
    const float* pred = (const float*)d_in[0];
    const int* target = (const int*)d_in[1];
    float* out = (float*)d_out;
    char* ws = (char*)d_ws;

    // workspace layout (all offsets NVOX-aligned)
    unsigned char* mask = (unsigned char*)ws;                               // 12 MiB
    unsigned char* ero  = (unsigned char*)(ws + (size_t)12 * NVOX);         // 12 MiB
    int* lblP           = (int*)(ws + (size_t)24 * NVOX);                   // 48 MiB
    int* lblQ           = (int*)(ws + (size_t)72 * NVOX);                   // 48 MiB
    unsigned char* reP  = (unsigned char*)(ws + (size_t)120 * NVOX);        // 12 MiB
    unsigned char* reQ  = (unsigned char*)(ws + (size_t)132 * NVOX);        // 12 MiB
    int* stats          = (int*)(ws + (size_t)144 * NVOX);                  // 144 B

    const dim3 blk(256);
    const dim3 tile_grid(WW / TX, HH / TY, (DD / TZ) * NMASK);  // (8, 8, 192)
    const dim3 red_grid(32, NMASK);                             // 256 waves / mask set

    zero_stats<<<1, 64, 0, stream>>>(stats);
    mask_kernel<<<(2 * NVOX) / 256, blk, 0, stream>>>(pred, target, mask);
    erode_kernel<<<(12 * NVOX) / 256, blk, 0, stream>>>(mask, ero);

    // connected components on masks
    lbl_init<<<(12 * NVOX) / 256, blk, 0, stream>>>(mask, lblP);
    for (int i = 0; i < ITERS_CC; ++i) {
        cc_step<<<tile_grid, blk, 0, stream>>>(lblP, lblQ, mask);
        int* t = lblP; lblP = lblQ; lblQ = t;
    }
    reduce_roots<<<red_grid, blk, 0, stream>>>(lblP, mask, stats);          // b0

    // connected components on eroded masks (reuse label buffers)
    lbl_init<<<(12 * NVOX) / 256, blk, 0, stream>>>(ero, lblP);
    for (int i = 0; i < ITERS_CC; ++i) {
        cc_step<<<tile_grid, blk, 0, stream>>>(lblP, lblQ, ero);
        int* t = lblP; lblP = lblQ; lblQ = t;
    }
    reduce_roots<<<red_grid, blk, 0, stream>>>(lblP, ero, stats + 12);      // b0e

    // background flood fill from border -> cavity volume
    flood_init<<<(12 * NVOX) / 256, blk, 0, stream>>>(mask, reP);
    for (int i = 0; i < ITERS_FL; ++i) {
        flood_step<<<tile_grid, blk, 0, stream>>>(reP, reQ, mask);
        unsigned char* t = reP; reP = reQ; reQ = t;
    }
    reduce_cavity<<<red_grid, blk, 0, stream>>>(reP, mask, stats + 24);     // cav

    finalize<<<1, 32, 0, stream>>>(stats, out);
}
